// SelfAttention_38182259261745
// MI455X (gfx1250) — matrix operand order repaired and checked by reading, not compiled, not run on hardware
//
#include <hip/hip_runtime.h>

#define Bq 4
#define Tt 2048
#define Cc 1024
#define Hh 16
#define Dd 64

typedef __attribute__((ext_vector_type(16))) __bf16 v16bf;
typedef __attribute__((ext_vector_type(8)))  float  v8f;

#if defined(__has_builtin)
#if __has_builtin(__builtin_amdgcn_global_load_async_to_lds_b128) && \
    __has_builtin(__builtin_amdgcn_s_wait_asynccnt)
#define HAS_ASYNC 1
#endif
#endif

#define AS1 __attribute__((address_space(1)))
#define AS3 __attribute__((address_space(3)))

typedef int v4i_gcc __attribute__((__vector_size__(4 * sizeof(int))));

static __device__ __forceinline__ void async_cp_b128(const void* g, void* l) {
#ifdef HAS_ASYNC
  __builtin_amdgcn_global_load_async_to_lds_b128(
      (AS1 v4i_gcc*)g, (AS3 v4i_gcc*)l, 0, 0);
#else
  *(uint4*)l = *(const uint4*)g;
#endif
}
static __device__ __forceinline__ void async_wait0() {
#ifdef HAS_ASYNC
  __builtin_amdgcn_s_wait_asynccnt(0);
#endif
}

static __device__ __forceinline__ unsigned short f2bf(float f) {
  unsigned int u = __float_as_uint(f);
  u += 0x7FFFu + ((u >> 16) & 1u);        // round-to-nearest-even
  return (unsigned short)(u >> 16);
}

// A-matrix (16x32 bf16) lane layout: lanes 0-15 rows M=0..15 hold K 0..7 (v0..v3)
// and 16..23 (v4..v7); lanes 16-31 same rows, K +8.
static __device__ __forceinline__ int kbaseA(int j, int hi) {
  return (j < 4 ? 2 * j : 16 + 2 * (j - 4)) + (hi ? 8 : 0);
}
// B-matrix (32x16 bf16) lane layout: lanes 0-15 N=lane hold K 0..7 (v0..v3)
// and 16..23 (v4..v7); lanes 16-31 N=lane-16, K +8.
static __device__ __forceinline__ int kbaseB(int j, int hi) {
  return (j < 4 ? 2 * j : 16 + 2 * (j - 4)) + (hi ? 8 : 0);
}

static __device__ __forceinline__ v8f wmma_bf16(const unsigned int a[8],
                                                const unsigned int b[8], v8f c) {
  union { v16bf v; unsigned int u[8]; } A, B;
#pragma unroll
  for (int i = 0; i < 8; ++i) { A.u[i] = a[i]; B.u[i] = b[i]; }
  return __builtin_amdgcn_wmma_f32_16x16x32_bf16(false, A.v, false, B.v,
                                                 (short)0, c, false, false);
}

// ---------------- fp32 -> bf16 conversion, 4 elements/thread ----------------
__global__ void cvt_bf16x4(const float* __restrict__ src,
                           unsigned short* __restrict__ dst, int n4) {
  int i = blockIdx.x * blockDim.x + threadIdx.x;
  if (i >= n4) return;
  float4 f = ((const float4*)src)[i];
  union { ushort4 s; uint2 u; } o;
  o.s.x = f2bf(f.x); o.s.y = f2bf(f.y); o.s.z = f2bf(f.z); o.s.w = f2bf(f.w);
  ((uint2*)dst)[i] = o.u;
}

// Stage one 64x32 A tile (async global->LDS) and one 32x64 W tile (transposed
// through VGPRs into Wt[n][k]) for the given K step.
static __device__ __forceinline__ void stage_tiles(
    const unsigned short* __restrict__ A, const unsigned short* __restrict__ W,
    int N, int K, int m0, int n0, int k0, int tid,
    unsigned short* Asb, unsigned short* Wtb) {
  {   // A tile 64x32 row-major: 128 threads x 32B, direct to LDS
    int row = tid >> 1, half = tid & 1;
    const unsigned short* g = A + (size_t)(m0 + row) * K + k0 + half * 16;
    unsigned short* l = Asb + row * 32 + half * 16;
    async_cp_b128(g, l);
    async_cp_b128(g + 8, l + 8);
  }
#pragma unroll
  for (int it = 0; it < 2; ++it) {   // W tile 32x64 -> transposed Wt[n][k]
    int idx = tid + it * 128;
    int krow = idx >> 3, seg = idx & 7;
    union { uint4 q; unsigned short s[8]; } u;
    u.q = *(const uint4*)(W + (size_t)(k0 + krow) * N + n0 + seg * 8);
#pragma unroll
    for (int e = 0; e < 8; ++e) Wtb[(seg * 8 + e) * 32 + krow] = u.s[e];
  }
}

// ---------------- bf16 GEMM with bias: out = A[M,K] @ W[K,N] + bias ---------
// mode 0: bf16 out permuted into [B,H,T,D] (QKV). mode 2: fp32 row-major out.
__global__ __launch_bounds__(128) void gemm_bf16(
    const unsigned short* __restrict__ A, const unsigned short* __restrict__ W,
    const float* __restrict__ bias, void* __restrict__ outp,
    int M, int N, int K, int mode) {
  __shared__ unsigned short As[2][64 * 32];   // ping-pong [row][k]
  __shared__ unsigned short Wt[2][64 * 32];   // ping-pong transposed [n][k]
  const int tid  = threadIdx.x;
  const int wave = tid >> 5;
  const int lane = tid & 31;
  const int hi   = lane >> 4;
  const int ln   = lane & 15;
  const int m0 = blockIdx.y * 64;
  const int n0 = blockIdx.x * 64;

  v8f acc[4];
#pragma unroll
  for (int i = 0; i < 4; ++i)
#pragma unroll
    for (int r = 0; r < 8; ++r) acc[i][r] = 0.0f;

  stage_tiles(A, W, N, K, m0, n0, 0, tid, As[0], Wt[0]);

  for (int k0 = 0; k0 < K; k0 += 32) {
    const int cur = (k0 >> 5) & 1;
    async_wait0();                       // own async copies landed in LDS
    __syncthreads();                     // everyone's copies visible
    if (k0 + 32 < K)                     // overlap next tile copy with WMMAs
      stage_tiles(A, W, N, K, m0, n0, k0 + 32, tid, As[cur ^ 1], Wt[cur ^ 1]);

    unsigned int afr[8];
#pragma unroll
    for (int j = 0; j < 8; ++j)
      afr[j] = *(const unsigned int*)(As[cur] + (wave * 16 + ln) * 32 + kbaseA(j, hi));
#pragma unroll
    for (int nt = 0; nt < 4; ++nt) {
      unsigned int bfr[8];
#pragma unroll
      for (int j = 0; j < 8; ++j)
        bfr[j] = *(const unsigned int*)(Wt[cur] + (nt * 16 + ln) * 32 + kbaseB(j, hi));
      acc[nt] = wmma_bf16(afr, bfr, acc[nt]);
    }
  }

#pragma unroll
  for (int nt = 0; nt < 4; ++nt) {
#pragma unroll
    for (int r = 0; r < 8; ++r) {
      int gm = m0 + wave * 16 + hi * 8 + r;      // C/D layout: M = r (+8 upper half)
      int gn = n0 + nt * 16 + ln;                // N = lane
      float v = acc[nt][r] + bias[gn];
      if (mode == 0) {
        int b = gm >> 11, t = gm & (Tt - 1);
        int h = gn >> 6,  d = gn & (Dd - 1);
        ((unsigned short*)outp)[(((size_t)(b * Hh + h)) * Tt + t) * Dd + d] = f2bf(v);
      } else {
        ((float*)outp)[(size_t)gm * N + gn] = v;
      }
    }
  }
}

// ---------------- flash attention: one wave per (b,h, 16-query tile) --------
__global__ __launch_bounds__(32) void attn_fwd(
    const unsigned short* __restrict__ Qm, const unsigned short* __restrict__ Km,
    const unsigned short* __restrict__ Vm, unsigned short* __restrict__ Ym) {
  __shared__ float Sld[16 * 32];
  __shared__ unsigned short Pld[16 * 32];
  __shared__ float mrow[16], lrow[16], arow[16];

  const int bh = blockIdx.x >> 7;      // T/16 = 128 query tiles
  const int qt = blockIdx.x & 127;
  const int q0 = qt * 16;
  const int lane = threadIdx.x;
  const int hi = lane >> 4;
  const int ln = lane & 15;

  const unsigned short* Qp = Qm + (size_t)bh * Tt * Dd;
  const unsigned short* Kp = Km + (size_t)bh * Tt * Dd;
  const unsigned short* Vp = Vm + (size_t)bh * Tt * Dd;

  // Q fragments: 16 rows x 64 d, two k=32 steps, A layout
  unsigned int qf[2][8];
#pragma unroll
  for (int f = 0; f < 2; ++f)
#pragma unroll
    for (int j = 0; j < 8; ++j)
      qf[f][j] = *(const unsigned int*)(Qp + (size_t)(q0 + ln) * Dd + f * 32 + kbaseA(j, hi));

  v8f o[4];
#pragma unroll
  for (int i = 0; i < 4; ++i)
#pragma unroll
    for (int r = 0; r < 8; ++r) o[i][r] = 0.0f;
  if (lane < 16) { mrow[lane] = -1e30f; lrow[lane] = 0.0f; }
  __syncthreads();

  for (int s0 = 0; s0 < q0 + 16; s0 += 32) {     // 32-key blocks, causal bound
    if (s0 + 32 < q0 + 16) {                     // prefetch next K/V block (L2 hot)
      __builtin_prefetch(Kp + (size_t)(s0 + 32 + lane) * Dd, 0, 3);
      __builtin_prefetch(Vp + (size_t)(s0 + 32 + lane) * Dd, 0, 3);
    }
#pragma unroll
    for (int kt = 0; kt < 2; ++kt) {             // S = Q @ K^T (scaled later)
      v8f s;
#pragma unroll
      for (int r = 0; r < 8; ++r) s[r] = 0.0f;
#pragma unroll
      for (int f = 0; f < 2; ++f) {
        unsigned int kf[8];
#pragma unroll
        for (int j = 0; j < 8; ++j)
          kf[j] = *(const unsigned int*)(Kp + (size_t)(s0 + kt * 16 + ln) * Dd +
                                         f * 32 + kbaseB(j, hi));
        s = wmma_bf16(qf[f], kf, s);
      }
#pragma unroll
      for (int r = 0; r < 8; ++r)
        Sld[(hi * 8 + r) * 32 + kt * 16 + ln] = s[r];
    }
    __syncthreads();

    if (lane < 16) {                              // online softmax per row
      const int r = lane, q = q0 + r;
      float m = mrow[r], bm = m;
      for (int c = 0; c < 32; ++c) {
        float sv = Sld[r * 32 + c] * 0.125f;      // 1/sqrt(64)
        if (s0 + c > q) sv = -1e30f;
        bm = fmaxf(bm, sv);
      }
      float alpha = __expf(m - bm);
      float ssum = 0.0f;
      for (int c = 0; c < 32; ++c) {
        float sv = Sld[r * 32 + c] * 0.125f;
        if (s0 + c > q) sv = -1e30f;
        float p = __expf(sv - bm);
        ssum += p;
        Pld[r * 32 + c] = f2bf(p);
      }
      mrow[r] = bm;
      lrow[r] = lrow[r] * alpha + ssum;
      arow[r] = alpha;
    }
    __syncthreads();

    float al[8];
#pragma unroll
    for (int r = 0; r < 8; ++r) al[r] = arow[hi * 8 + r];
    unsigned int pf[8];
#pragma unroll
    for (int j = 0; j < 8; ++j)
      pf[j] = *(const unsigned int*)(Pld + ln * 32 + kbaseA(j, hi));
#pragma unroll
    for (int nt = 0; nt < 4; ++nt) {              // O = O*alpha + P @ V
#pragma unroll
      for (int r = 0; r < 8; ++r) o[nt][r] *= al[r];
      unsigned int vf[8];
#pragma unroll
      for (int j = 0; j < 8; ++j) {
        int k = kbaseB(j, hi);
        int dcol = nt * 16 + ln;
        unsigned int lo  = Vp[(size_t)(s0 + k)     * Dd + dcol];
        unsigned int hi2 = Vp[(size_t)(s0 + k + 1) * Dd + dcol];
        vf[j] = lo | (hi2 << 16);
      }
      o[nt] = wmma_bf16(pf, vf, o[nt]);
    }
    __syncthreads();
  }

  float linv[8];
#pragma unroll
  for (int r = 0; r < 8; ++r) linv[r] = 1.0f / lrow[hi * 8 + r];
  const int b = bh >> 4, h = bh & 15;
#pragma unroll
  for (int nt = 0; nt < 4; ++nt)
#pragma unroll
    for (int r = 0; r < 8; ++r) {
      int m = q0 + hi * 8 + r;
      int d = nt * 16 + ln;
      Ym[((size_t)(b * Tt + m)) * Cc + h * Dd + d] = f2bf(o[nt][r] * linv[r]);
    }
}

// ---------------------------------------------------------------------------
extern "C" void kernel_launch(void* const* d_in, const int* in_sizes, int n_in,
                              void* d_out, int out_size, void* d_ws, size_t ws_size,
                              hipStream_t stream) {
  (void)in_sizes; (void)n_in; (void)out_size; (void)ws_size;
  const float* x  = (const float*)d_in[0];
  const float* Wk = (const float*)d_in[1];
  const float* bk = (const float*)d_in[2];
  const float* Wq = (const float*)d_in[3];
  const float* bq = (const float*)d_in[4];
  const float* Wv = (const float*)d_in[5];
  const float* bv = (const float*)d_in[6];
  const float* Wp = (const float*)d_in[7];
  const float* bp = (const float*)d_in[8];

  unsigned char* ws = (unsigned char*)d_ws;
  const size_t MB = 1024 * 1024;
  unsigned short* xb  = (unsigned short*)(ws +  0 * MB);   // 16 MiB: x in bf16
  unsigned short* wkb = (unsigned short*)(ws + 16 * MB);   //  2 MiB each
  unsigned short* wqb = (unsigned short*)(ws + 18 * MB);
  unsigned short* wvb = (unsigned short*)(ws + 20 * MB);
  unsigned short* wpb = (unsigned short*)(ws + 22 * MB);
  unsigned short* Qb  = (unsigned short*)(ws + 24 * MB);   // 16 MiB [B,H,T,D]
  unsigned short* Kb  = (unsigned short*)(ws + 40 * MB);
  unsigned short* Vb  = (unsigned short*)(ws + 56 * MB);
  unsigned short* Yb  = (unsigned short*)(ws + 72 * MB);   // 16 MiB [B*T, C]

  const int nx4 = (Bq * Tt * Cc) / 4;
  const int nw4 = (Cc * Cc) / 4;
  cvt_bf16x4<<<(nx4 + 255) / 256, 256, 0, stream>>>(x,  xb,  nx4);
  cvt_bf16x4<<<(nw4 + 255) / 256, 256, 0, stream>>>(Wk, wkb, nw4);
  cvt_bf16x4<<<(nw4 + 255) / 256, 256, 0, stream>>>(Wq, wqb, nw4);
  cvt_bf16x4<<<(nw4 + 255) / 256, 256, 0, stream>>>(Wv, wvb, nw4);
  cvt_bf16x4<<<(nw4 + 255) / 256, 256, 0, stream>>>(Wp, wpb, nw4);

  dim3 gg(Cc / 64, (Bq * Tt) / 64);   // (16, 128) blocks of 128 threads
  gemm_bf16<<<gg, 128, 0, stream>>>(xb, wkb, bk, Kb, Bq * Tt, Cc, Cc, 0);
  gemm_bf16<<<gg, 128, 0, stream>>>(xb, wqb, bq, Qb, Bq * Tt, Cc, Cc, 0);
  gemm_bf16<<<gg, 128, 0, stream>>>(xb, wvb, bv, Vb, Bq * Tt, Cc, Cc, 0);

  attn_fwd<<<Bq * Hh * (Tt / 16), 32, 0, stream>>>(Qb, Kb, Vb, Yb);

  gemm_bf16<<<gg, 128, 0, stream>>>(Yb, wpb, bp, d_out, Bq * Tt, Cc, Cc, 2);
}
